// Attention_70411693851043
// MI455X (gfx1250) — compile-verified, matrix operand order checked by reading, not run on hardware
//
#include <hip/hip_runtime.h>
#include <hip/hip_bf16.h>

// Problem dims (fixed by the reference)
#define BB 2
#define SS 2048
#define DD 2048
#define HH 16
#define HD 128

typedef __bf16 bf16;
typedef __attribute__((ext_vector_type(16))) __bf16 v16bf;
typedef __attribute__((ext_vector_type(8)))  __bf16 v8bf;
typedef __attribute__((ext_vector_type(8)))  float  v8f;

// ---------------------------------------------------------------------------
// WMMA helpers (CDNA5 wave32, 16x16x32 bf16 -> f32)
// ---------------------------------------------------------------------------
__device__ __forceinline__ v8f wmma_bf16(v16bf a, v16bf b, v8f c) {
  // 8 args: (neg_a, A, neg_b, B, c_mod, C, reuse_a, reuse_b)
  return __builtin_amdgcn_wmma_f32_16x16x32_bf16(false, a, false, b,
                                                 (short)0, c, false, false);
}

// Load a 16x32 bf16 A-fragment (or symmetric B-fragment) from a row-major
// tile whose 16 "rows" have stride `ld` elements.
// Per CDNA5 ISA layout: lanes 0-15 hold row M=lane with K=0..7 and K=16..23;
// lanes 16-31 hold row M=lane-16 with K=8..15 and K=24..31.
__device__ __forceinline__ v16bf load_frag(const bf16* __restrict__ base, int ld) {
  const int lane = threadIdx.x & 31;
  const int r    = lane & 15;
  const int kb   = (lane >> 4) << 3;           // 0 or 8
  const bf16* p  = base + (size_t)r * ld + kb;
  v8bf lo = *(const v8bf*)(p);                 // K = kb .. kb+7
  v8bf hi = *(const v8bf*)(p + 16);            // K = kb+16 .. kb+23
  v16bf out;
#pragma unroll
  for (int i = 0; i < 8; ++i) { out[i] = lo[i]; out[i + 8] = hi[i]; }
  return out;
}

// Cross-lane reductions within each 16-lane half (matches C-layout rows)
__device__ __forceinline__ float rmax16(float v) {
#pragma unroll
  for (int m = 1; m < 16; m <<= 1) v = fmaxf(v, __shfl_xor(v, m, 32));
  return v;
}
__device__ __forceinline__ float rsum16(float v) {
#pragma unroll
  for (int m = 1; m < 16; m <<= 1) v += __shfl_xor(v, m, 32);
  return v;
}

// ---------------------------------------------------------------------------
// Elementwise kernels
// ---------------------------------------------------------------------------
__global__ void cast_f32_to_bf16(const float* __restrict__ in,
                                 bf16* __restrict__ out, int n) {
  int idx = blockIdx.x * blockDim.x + threadIdx.x;
  if (idx < n) out[idx] = (bf16)in[idx];
}

// lin [B,S,D] (bf16) -> rope + reshape -> out [B,H,S,HD] (bf16)
__global__ void rope_reshape(const bf16* __restrict__ lin,
                             const float* __restrict__ fc,
                             const float* __restrict__ fs,
                             bf16* __restrict__ outp) {
  const int NP = BB * SS * HH * (HD / 2);
  int idx = blockIdx.x * blockDim.x + threadIdx.x;
  if (idx >= NP) return;
  int i   = idx % (HD / 2);
  int t   = idx / (HD / 2);
  int h   = t % HH;  t /= HH;
  int s   = t % SS;
  int b   = t / SS;
  float c  = fc[s * (HD / 2) + i];
  float sn = fs[s * (HD / 2) + i];
  size_t src = ((size_t)(b * SS + s)) * DD + h * HD + 2 * i;
  float xr = (float)lin[src], xi = (float)lin[src + 1];
  size_t dst = (((size_t)(b * HH + h)) * SS + s) * HD + 2 * i;
  outp[dst]     = (bf16)(xr * c - xi * sn);
  outp[dst + 1] = (bf16)(xr * sn + xi * c);
}

// lin [B,S,D] (bf16) -> vt [B,H,HD,S] (bf16)  (transposed so PV B-frags are contiguous)
__global__ void v_transpose(const bf16* __restrict__ lin, bf16* __restrict__ vt) {
  const int N = BB * SS * DD;
  int idx = blockIdx.x * blockDim.x + threadIdx.x;
  if (idx >= N) return;
  int d = idx % DD;
  int t = idx / DD;
  int s = t % SS;
  int b = t / SS;
  int h = d / HD, hd = d % HD;
  vt[(((size_t)(b * HH + h)) * HD + hd) * SS + s] = lin[idx];
}

// ---------------------------------------------------------------------------
// NT GEMM: C[m,n] = sum_k A[m,k] * W[n,k]   (A: MxK row-major, W: NxK row-major)
// One wave = 32x64 output tile (2x4 WMMA accumulators): per 32-deep k-step it
// loads 2 A-frags + 4 B-frags (6 KB) and issues 8 WMMAs -> ~21 FLOP/B, with
// every B-fragment reused across two row-tiles in registers.
// Block = 4 waves = 32x256 output.
// ---------------------------------------------------------------------------
template <int STORE_F32>
__global__ __launch_bounds__(128) void gemm_nt(const bf16* __restrict__ A,
                                               const bf16* __restrict__ W,
                                               float* __restrict__ Cf,
                                               bf16* __restrict__ Cb,
                                               int M, int N, int K) {
  const int wave = threadIdx.x >> 5;
  const int lane = threadIdx.x & 31;
  const int m0 = blockIdx.y * 32;
  const int n0 = blockIdx.x * 256 + wave * 64;

  v8f acc0[4] = {};
  v8f acc1[4] = {};
  for (int k0 = 0; k0 < K; k0 += 32) {
    v16bf a0 = load_frag(A + (size_t)m0 * K + k0, K);
    v16bf a1 = load_frag(A + (size_t)(m0 + 16) * K + k0, K);
#pragma unroll
    for (int t = 0; t < 4; ++t) {
      v16bf b = load_frag(W + (size_t)(n0 + t * 16) * K + k0, K);
      acc0[t] = wmma_bf16(a0, b, acc0[t]);
      acc1[t] = wmma_bf16(a1, b, acc1[t]);
    }
  }

  const int half = lane >> 4;
  const int nn   = lane & 15;
#pragma unroll
  for (int t = 0; t < 4; ++t) {
#pragma unroll
    for (int i = 0; i < 8; ++i) {
      size_t m = (size_t)(m0 + i + 8 * half);
      size_t n = (size_t)(n0 + t * 16 + nn);
      if (STORE_F32) {
        Cf[m * N + n]        = acc0[t][i];
        Cf[(m + 16) * N + n] = acc1[t][i];
      } else {
        Cb[m * N + n]        = (bf16)acc0[t][i];
        Cb[(m + 16) * N + n] = (bf16)acc1[t][i];
      }
    }
  }
}

// ---------------------------------------------------------------------------
// Flash attention (causal, start_pos = 0).
// Q,K: [B,H,S,HD] bf16 ; Vt: [B,H,HD,S] bf16 ; O: [B,S,D] bf16
// One wave handles a 16-query block; online softmax over 32-key blocks.
// ---------------------------------------------------------------------------
__global__ __launch_bounds__(128) void attn_fwd(const bf16* __restrict__ Q,
                                                const bf16* __restrict__ Kt,
                                                const bf16* __restrict__ Vt,
                                                bf16* __restrict__ O) {
  __shared__ __align__(16) bf16 plds[4][16 * 32];
  const int wave = threadIdx.x >> 5;
  const int lane = threadIdx.x & 31;
  const int q0 = blockIdx.x * 64 + wave * 16;
  const int h  = blockIdx.y;
  const int b  = blockIdx.z;
  const size_t bh = (size_t)(b * HH + h);

  const bf16* qb = Q  + (bh * SS + q0) * HD;
  const bf16* kb = Kt + bh * SS * HD;
  const bf16* vb = Vt + bh * HD * SS;

  v16bf qf[4];
#pragma unroll
  for (int c = 0; c < 4; ++c) qf[c] = load_frag(qb + c * 32, HD);

  v8f o[8] = {};
  float mrun[8], lrun[8];
#pragma unroll
  for (int i = 0; i < 8; ++i) { mrun[i] = -3.0e38f; lrun[i] = 0.0f; }

  const int half = lane >> 4;
  const int nn   = lane & 15;
  const float scale = 0.08838834764831845f;  // 1/sqrt(128)

  for (int j0 = 0; j0 <= q0 + 15; j0 += 32) {
    // ---- scores: two 16x16 tiles over HD=128 (4 k-chunks of 32) ----
    v8f s0 = {}, s1 = {};
#pragma unroll
    for (int c = 0; c < 4; ++c) {
      v16bf k0f = load_frag(kb + (size_t)j0 * HD + c * 32, HD);
      v16bf k1f = load_frag(kb + (size_t)(j0 + 16) * HD + c * 32, HD);
      s0 = wmma_bf16(qf[c], k0f, s0);
      s1 = wmma_bf16(qf[c], k1f, s1);
    }

    // ---- online softmax update (per row m = i + 8*half) ----
    float p0[8], p1[8];
#pragma unroll
    for (int i = 0; i < 8; ++i) {
      int q = q0 + i + 8 * half;
      float v0 = s0[i] * scale + ((j0 + nn)      > q ? -1.0e9f : 0.0f);
      float v1 = s1[i] * scale + ((j0 + 16 + nn) > q ? -1.0e9f : 0.0f);
      float rm   = rmax16(fmaxf(v0, v1));
      float mnew = fmaxf(mrun[i], rm);
      float fac  = __expf(mrun[i] - mnew);
      float e0 = __expf(v0 - mnew);
      float e1 = __expf(v1 - mnew);
      float rs = rsum16(e0 + e1);
      lrun[i] = lrun[i] * fac + rs;
      mrun[i] = mnew;
      p0[i] = e0; p1[i] = e1;
#pragma unroll
      for (int t = 0; t < 8; ++t) o[t][i] *= fac;
    }

    // ---- C-layout -> A-fragment via LDS ----
    bf16* pl = plds[wave];
#pragma unroll
    for (int i = 0; i < 8; ++i) {
      int m = i + 8 * half;
      pl[m * 32 + nn]      = (bf16)p0[i];
      pl[m * 32 + 16 + nn] = (bf16)p1[i];
    }
    asm volatile("s_wait_dscnt 0x0" ::: "memory");
    v16bf pf = load_frag(pl, 32);

    // ---- PV: 8 output tiles covering HD=128 ----
#pragma unroll
    for (int t = 0; t < 8; ++t) {
      v16bf vf = load_frag(vb + (size_t)(t * 16) * SS + j0, SS);
      o[t] = wmma_bf16(pf, vf, o[t]);
    }
  }

  // ---- normalize and store to [B,S,D] bf16 ----
#pragma unroll
  for (int t = 0; t < 8; ++t) {
#pragma unroll
    for (int i = 0; i < 8; ++i) {
      int q = q0 + i + 8 * half;
      float val = o[t][i] / lrun[i];
      O[((size_t)(b * SS + q)) * DD + h * HD + t * 16 + nn] = (bf16)val;
    }
  }
}

// ---------------------------------------------------------------------------
// Host-side launch
// ---------------------------------------------------------------------------
extern "C" void kernel_launch(void* const* d_in, const int* in_sizes, int n_in,
                              void* d_out, int out_size, void* d_ws, size_t ws_size,
                              hipStream_t stream) {
  (void)in_sizes; (void)n_in; (void)out_size; (void)ws_size;
  const float* x  = (const float*)d_in[0];
  const float* wq = (const float*)d_in[1];
  const float* wk = (const float*)d_in[2];
  const float* wv = (const float*)d_in[3];
  const float* wo = (const float*)d_in[4];
  const float* fc = (const float*)d_in[5];
  const float* fs = (const float*)d_in[6];
  // d_in[7] = mask (causal -1e9 triu, applied analytically), d_in[8] = start_pos (0)

  const size_t NX = (size_t)BB * SS * DD;  // 8,388,608 activations
  const size_t NW = (size_t)DD * DD;       // 4,194,304 per weight

  char* ws = (char*)d_ws;
  bf16* xb   = (bf16*)ws; ws += NX * 2;
  bf16* wqb  = (bf16*)ws; ws += NW * 2;
  bf16* wkb  = (bf16*)ws; ws += NW * 2;
  bf16* wvb  = (bf16*)ws; ws += NW * 2;
  bf16* wob  = (bf16*)ws; ws += NW * 2;
  bf16* qlin = (bf16*)ws; ws += NX * 2;
  bf16* klin = (bf16*)ws; ws += NX * 2;
  bf16* vlin = (bf16*)ws; ws += NX * 2;
  bf16* qr   = (bf16*)ws; ws += NX * 2;
  bf16* kr   = (bf16*)ws; ws += NX * 2;
  bf16* vt   = (bf16*)ws; ws += NX * 2;
  bf16* ao   = (bf16*)ws; ws += NX * 2;   // total ~160 MiB

  // 1) downconvert inputs to bf16
  cast_f32_to_bf16<<<(int)((NX + 255) / 256), 256, 0, stream>>>(x,  xb,  (int)NX);
  cast_f32_to_bf16<<<(int)((NW + 255) / 256), 256, 0, stream>>>(wq, wqb, (int)NW);
  cast_f32_to_bf16<<<(int)((NW + 255) / 256), 256, 0, stream>>>(wk, wkb, (int)NW);
  cast_f32_to_bf16<<<(int)((NW + 255) / 256), 256, 0, stream>>>(wv, wvb, (int)NW);
  cast_f32_to_bf16<<<(int)((NW + 255) / 256), 256, 0, stream>>>(wo, wob, (int)NW);

  // 2) QKV projections: [4096 x 2048] = x * W^T   (NT layout, WMMA)
  dim3 gg(DD / 256, (BB * SS) / 32);
  gemm_nt<0><<<gg, 128, 0, stream>>>(xb, wqb, nullptr, qlin, BB * SS, DD, DD);
  gemm_nt<0><<<gg, 128, 0, stream>>>(xb, wkb, nullptr, klin, BB * SS, DD, DD);
  gemm_nt<0><<<gg, 128, 0, stream>>>(xb, wvb, nullptr, vlin, BB * SS, DD, DD);

  // 3) RoPE + head-major reshape for Q/K; transpose V
  const int NP = BB * SS * HH * (HD / 2);
  rope_reshape<<<(NP + 255) / 256, 256, 0, stream>>>(qlin, fc, fs, qr);
  rope_reshape<<<(NP + 255) / 256, 256, 0, stream>>>(klin, fc, fs, kr);
  v_transpose<<<(int)((NX + 255) / 256), 256, 0, stream>>>(vlin, vt);

  // 4) causal flash attention
  dim3 ga(SS / 64, HH, BB);
  attn_fwd<<<ga, 128, 0, stream>>>(qr, kr, vt, ao);

  // 5) output projection -> fp32 d_out
  gemm_nt<1><<<gg, 128, 0, stream>>>(ao, wob, (float*)d_out, nullptr, BB * SS, DD, DD);
}
